// SimpleAttention_85117661872164
// MI455X (gfx1250) — matrix operand order repaired and checked by reading, not compiled, not run on hardware
//
#include <hip/hip_runtime.h>

// ---------------------------------------------------------------------------
// Causal self-attention with rotary embeddings, bf16 WMMA pipeline (gfx1250).
// B=2, S=2048, D=2048, H=16, dk=128.
// GEMMs: async-to-LDS double-buffered tiles (BK=64) feeding v_wmma bf16.
// ---------------------------------------------------------------------------

#define B_   2
#define S_   2048
#define D_   2048
#define H_   16
#define DK_  128
#define E3_  (3 * D_)   // 6144

typedef __attribute__((ext_vector_type(16))) __bf16 bf16x16;
typedef __attribute__((ext_vector_type(8)))  __bf16 bf16x8;
typedef __attribute__((ext_vector_type(8)))  float  floatx8;

__device__ __forceinline__ __bf16 f2bf(float f) {
    unsigned u = __builtin_bit_cast(unsigned, f);
    unsigned r = (u + 0x7FFFu + ((u >> 16) & 1u)) >> 16;
    unsigned short h = (unsigned short)r;
    return __builtin_bit_cast(__bf16, h);
}

__device__ __forceinline__ bf16x16 pack16(bf16x8 lo, bf16x8 hi) {
    bf16x16 r;
#pragma unroll
    for (int i = 0; i < 8; ++i) { r[i] = lo[i]; r[i + 8] = hi[i]; }
    return r;
}

// Async copy of 16 bytes global -> LDS (gfx1250, GV mode, tracked by ASYNCcnt).
__device__ __forceinline__ void async_copy_b128(const void* gptr, void* lptr) {
    unsigned lds32 = (unsigned)(unsigned long long)lptr;       // LDS byte address
    unsigned long long ga = (unsigned long long)gptr;          // 64-bit global addr
    asm volatile("global_load_async_to_lds_b128 %0, %1, off"
                 :: "v"(lds32), "v"(ga) : "memory");
}
__device__ __forceinline__ void wait_async0() {
    asm volatile("s_wait_asynccnt 0x0" ::: "memory");
}

// ---------------------------------------------------------------------------
// Kernel 1: fp32 -> bf16 cast (grid-stride)
// ---------------------------------------------------------------------------
__global__ void cvt_f32_bf16(const float* __restrict__ in, __bf16* __restrict__ out,
                             long long n) {
    long long stride = (long long)gridDim.x * blockDim.x;
    for (long long i = (long long)blockIdx.x * blockDim.x + threadIdx.x; i < n; i += stride)
        out[i] = f2bf(in[i]);
}

// ---------------------------------------------------------------------------
// Kernel 2: GEMM  C[M,N] = A[M,K] * B[N,K]^T   (bf16 in, fp32 out)
// Block = 256 threads = 8 waves. Block tile 128x128, BK=64, wave tile 32x64.
// Double-buffered LDS (2 x 32KB) staged via global_load_async_to_lds_b128;
// 16 wmma per barrier.
// ---------------------------------------------------------------------------
__global__ __launch_bounds__(256)
void gemm_bf16_nt(const __bf16* __restrict__ A, const __bf16* __restrict__ Bm,
                  float* __restrict__ C, int M, int N, int K) {
    __shared__ __align__(16) __bf16 As[2][128 * 64];
    __shared__ __align__(16) __bf16 Bs[2][128 * 64];

    const int tid  = threadIdx.x;
    const int lane = tid & 31;
    const int wave = tid >> 5;
    const int grp  = lane >> 4;
    const int ln   = lane & 15;
    const int wm   = wave >> 1;   // 0..3 -> 32-row strip
    const int wn   = wave & 1;    // 0..1 -> 64-col strip

    const int m0 = blockIdx.y * 128;
    const int n0 = blockIdx.x * 128;

    floatx8 acc[2][4];
#pragma unroll
    for (int i = 0; i < 2; ++i)
#pragma unroll
        for (int t = 0; t < 4; ++t) acc[i][t] = (floatx8)0.0f;

    // Stage one BK=64 slab of A (128x64) and B (128x64) into LDS buffer `buf`.
    // 1024 16B chunks each; every thread issues 4 A-chunks and 4 B-chunks —
    // no conditions, so no EXEC-mask divergence around the async ops.
    auto stage = [&](int buf, int kk) {
#pragma unroll
        for (int i = 0; i < 4; ++i) {
            int c   = tid + i * 256;       // 0..1023
            int row = c >> 3, seg = c & 7; // row 0..127, seg*8 bf16 within row
            async_copy_b128(A + (size_t)(m0 + row) * K + kk + seg * 8,
                            &As[buf][row * 64 + seg * 8]);
            async_copy_b128(Bm + (size_t)(n0 + row) * K + kk + seg * 8,
                            &Bs[buf][row * 64 + seg * 8]);
        }
    };

    const int ksteps = K >> 6;
    stage(0, 0);

    for (int kt = 0; kt < ksteps; ++kt) {
        const int buf = kt & 1;
        wait_async0();        // this wave's copies into `buf` landed in LDS
        __syncthreads();      // all waves' copies landed; all reads of buf^1 done
        if (kt + 1 < ksteps) stage(buf ^ 1, (kt + 1) * 64);

#pragma unroll
        for (int h2 = 0; h2 < 2; ++h2) {   // two K=32 sub-steps within the slab
            const int kk2 = h2 * 32;
            bf16x16 af[2];
#pragma unroll
            for (int im = 0; im < 2; ++im) {
                // A-frag (16-bit 16x32): g=0 holds K[0..7]&[16..23], g=1 K[8..15]&[24..31]
                const __bf16* pa =
                    &As[buf][(wm * 32 + im * 16 + ln) * 64 + kk2 + grp * 8];
                af[im] = pack16(*(const bf16x8*)pa, *(const bf16x8*)(pa + 16));
            }
            bf16x16 bf_[4];
#pragma unroll
            for (int t = 0; t < 4; ++t) {
                // B-frag (32x16): lane -> col n, g=0 holds K[0..7]&[16..23], g=1 K[8..15]&[24..31]
                const __bf16* pb =
                    &Bs[buf][(wn * 64 + t * 16 + ln) * 64 + kk2 + grp * 8];
                bf_[t] = pack16(*(const bf16x8*)pb, *(const bf16x8*)(pb + 16));
            }
#pragma unroll
            for (int im = 0; im < 2; ++im)
#pragma unroll
                for (int t = 0; t < 4; ++t)
                    acc[im][t] = __builtin_amdgcn_wmma_f32_16x16x32_bf16(
                        false, af[im], false, bf_[t], (short)0, acc[im][t], false, false);
        }
    }

#pragma unroll
    for (int im = 0; im < 2; ++im)
#pragma unroll
        for (int t = 0; t < 4; ++t)
#pragma unroll
            for (int r = 0; r < 8; ++r) {
                int row = m0 + wm * 32 + im * 16 + r + 8 * grp;
                int col = n0 + wn * 64 + t * 16 + ln;
                C[(size_t)row * N + col] = acc[im][t][r];
            }
}

// ---------------------------------------------------------------------------
// Kernel 3: rotary + split.  qkv fp32 [B,S,3*D] ->
//   Qb bf16 [B,H,S,dk] (scaled by 1/sqrt(dk)), Kb bf16 [B,H,S,dk],
//   Vt bf16 [B,H,dk,S] (transposed for contiguous PV B-fragments).
// ---------------------------------------------------------------------------
__global__ void rotary_split(const float* __restrict__ qkv,
                             __bf16* __restrict__ Qb, __bf16* __restrict__ Kb,
                             __bf16* __restrict__ Vt) {
    int idx = blockIdx.x * blockDim.x + threadIdx.x;      // [0, B*S*H*64)
    int i2 = idx & 63;  int t = idx >> 6;
    int h  = t & (H_ - 1); t >>= 4;
    int s  = t & (S_ - 1); int b = t >> 11;

    size_t base = (size_t)(b * S_ + s) * E3_ + h * DK_;
    float q1 = qkv[base + i2],        q2 = qkv[base + 64 + i2];
    float k1 = qkv[base + D_ + i2],   k2 = qkv[base + D_ + 64 + i2];
    float v1 = qkv[base + 2*D_ + i2], v2 = qkv[base + 2*D_ + 64 + i2];

    float freq  = __expf(-9.210340371976184f * ((float)i2 / 63.0f));  // 10000^(-i/63)
    float theta = (float)s * freq;
    float c = __cosf(theta), sn = __sinf(theta);
    const float qs = 0.08838834764831845f;  // 1/sqrt(128)

    size_t bh  = (size_t)(b * H_ + h);
    size_t qko = (bh * S_ + s) * DK_;
    Qb[qko + i2]      = f2bf((q1 * c - q2 * sn) * qs);
    Qb[qko + 64 + i2] = f2bf((q1 * sn + q2 * c) * qs);
    Kb[qko + i2]      = f2bf(k1 * c - k2 * sn);
    Kb[qko + 64 + i2] = f2bf(k1 * sn + k2 * c);
    Vt[(bh * DK_ + i2) * S_ + s]      = f2bf(v1);
    Vt[(bh * DK_ + 64 + i2) * S_ + s] = f2bf(v2);
}

// ---------------------------------------------------------------------------
// Kernel 4: flash attention (causal).  One wave per 16-row Q block.
// Per 32-key step: 8 wmma for S = Q*K^T, online softmax in registers,
// P transposed via LDS to A-layout, 4 wmma for O += P*V.
// ---------------------------------------------------------------------------
__global__ __launch_bounds__(128)
void flash_attn(const __bf16* __restrict__ Qb, const __bf16* __restrict__ Kb,
                const __bf16* __restrict__ Vt, __bf16* __restrict__ Ob) {
    __shared__ float plds[4][16 * 32];

    const int lane = threadIdx.x & 31;
    const int wave = threadIdx.x >> 5;
    const int grp  = lane >> 4;
    const int ln   = lane & 15;

    const int qblk = blockIdx.x * 4 + wave;
    const int h = blockIdx.y, b = blockIdx.z;
    const size_t bh = (size_t)(b * H_ + h);
    const int m0 = qblk * 16;

    const __bf16* Qp = Qb + bh * S_ * DK_;
    const __bf16* Kp = Kb + bh * S_ * DK_;
    const __bf16* Vp = Vt + bh * DK_ * S_;

    bf16x16 qf[4];
#pragma unroll
    for (int f = 0; f < 4; ++f) {
        const __bf16* p = Qp + (size_t)(m0 + ln) * DK_ + f * 32 + grp * 8;
        qf[f] = pack16(*(const bf16x8*)p, *(const bf16x8*)(p + 16));
    }

    floatx8 o[4];
#pragma unroll
    for (int t = 0; t < 4; ++t) o[t] = (floatx8)0.0f;
    float mr[8], lr[8];
#pragma unroll
    for (int r = 0; r < 8; ++r) { mr[r] = -3.0e38f; lr[r] = 0.0f; }

    const int jend = (m0 + 15) / 32;
    for (int j = 0; j <= jend; ++j) {
        const int c0 = j * 32;

        floatx8 sfr[2];
        sfr[0] = (floatx8)0.0f; sfr[1] = (floatx8)0.0f;
#pragma unroll
        for (int f = 0; f < 4; ++f) {
#pragma unroll
            for (int t = 0; t < 2; ++t) {
                const __bf16* p = Kp + (size_t)(c0 + t * 16 + ln) * DK_ + f * 32 + grp * 8;
                bf16x16 kf = pack16(*(const bf16x8*)p, *(const bf16x8*)(p + 16));
                sfr[t] = __builtin_amdgcn_wmma_f32_16x16x32_bf16(
                    false, qf[f], false, kf, (short)0, sfr[t], false, false);
            }
        }

#pragma unroll
        for (int t = 0; t < 2; ++t)
#pragma unroll
            for (int r = 0; r < 8; ++r) {
                int row = m0 + r + 8 * grp;
                int col = c0 + t * 16 + ln;
                if (col > row) sfr[t][r] = -3.0e38f;
            }

        float alpha[8];
#pragma unroll
        for (int r = 0; r < 8; ++r) {
            float v = fmaxf(sfr[0][r], sfr[1][r]);
#pragma unroll
            for (int off = 8; off >= 1; off >>= 1) v = fmaxf(v, __shfl_xor(v, off, 16));
            float mnew = fmaxf(mr[r], v);
            alpha[r] = __expf(mr[r] - mnew);
            mr[r] = mnew;
        }
#pragma unroll
        for (int t = 0; t < 2; ++t)
#pragma unroll
            for (int r = 0; r < 8; ++r) sfr[t][r] = __expf(sfr[t][r] - mr[r]);
#pragma unroll
        for (int r = 0; r < 8; ++r) {
            float rs = sfr[0][r] + sfr[1][r];
#pragma unroll
            for (int off = 8; off >= 1; off >>= 1) rs += __shfl_xor(rs, off, 16);
            lr[r] = lr[r] * alpha[r] + rs;
#pragma unroll
            for (int t = 0; t < 4; ++t) o[t][r] *= alpha[r];
        }

        float* pl = &plds[wave][0];
#pragma unroll
        for (int t = 0; t < 2; ++t)
#pragma unroll
            for (int r = 0; r < 8; ++r)
                pl[(r + 8 * grp) * 32 + t * 16 + ln] = sfr[t][r];
        __builtin_amdgcn_wave_barrier();

        bf16x16 pf;
#pragma unroll
        for (int e = 0; e < 16; ++e) {
            int k = grp * 16 + e;
            pf[e] = f2bf(pl[ln * 32 + k]);
        }
        __builtin_amdgcn_wave_barrier();

#pragma unroll
        for (int t = 0; t < 4; ++t) {
            const __bf16* p = Vp + (size_t)(t * 16 + ln) * S_ + c0 + grp * 16;
            bf16x16 vf = pack16(*(const bf16x8*)p, *(const bf16x8*)(p + 8));
            o[t] = __builtin_amdgcn_wmma_f32_16x16x32_bf16(
                false, pf, false, vf, (short)0, o[t], false, false);
        }
    }

#pragma unroll
    for (int t = 0; t < 4; ++t)
#pragma unroll
        for (int r = 0; r < 8; ++r) {
            int row = m0 + r + 8 * grp;
            int col = h * DK_ + t * 16 + ln;
            Ob[((size_t)b * S_ + row) * D_ + col] = f2bf(o[t][r] / lr[r]);
        }
}

// ---------------------------------------------------------------------------
// Host: pipeline launch
// ---------------------------------------------------------------------------
extern "C" void kernel_launch(void* const* d_in, const int* in_sizes, int n_in,
                              void* d_out, int out_size, void* d_ws, size_t ws_size,
                              hipStream_t stream) {
    const float* x     = (const float*)d_in[0];  // [B,S,D]
    const float* w_qkv = (const float*)d_in[1];  // [3D,D]
    const float* w_o   = (const float*)d_in[2];  // [D,D]

    const long long nX = (long long)B_ * S_ * D_;
    const long long nWqkv = (long long)E3_ * D_;
    const long long nWo = (long long)D_ * D_;

    char* p = (char*)d_ws;
    auto carve = [&](size_t bytes) -> void* {
        void* r = (void*)p;
        p += (bytes + 255) & ~(size_t)255;
        return r;
    };
    __bf16* xb    = (__bf16*)carve(nX * 2);
    __bf16* wqkvb = (__bf16*)carve(nWqkv * 2);
    __bf16* wob   = (__bf16*)carve(nWo * 2);
    float*  qkvf  = (float*) carve((size_t)B_ * S_ * E3_ * 4);
    __bf16* Qb    = (__bf16*)carve(nX * 2);
    __bf16* Kb    = (__bf16*)carve(nX * 2);
    __bf16* Vt    = (__bf16*)carve(nX * 2);
    __bf16* attnb = (__bf16*)carve(nX * 2);

    cvt_f32_bf16<<<2048, 256, 0, stream>>>(x, xb, nX);
    cvt_f32_bf16<<<2048, 256, 0, stream>>>(w_qkv, wqkvb, nWqkv);
    cvt_f32_bf16<<<2048, 256, 0, stream>>>(w_o, wob, nWo);

    // qkv = x @ w_qkv^T : M=4096, N=6144, K=2048
    gemm_bf16_nt<<<dim3(E3_ / 128, (B_ * S_) / 128), 256, 0, stream>>>(
        xb, wqkvb, qkvf, B_ * S_, E3_, D_);

    rotary_split<<<(B_ * S_ * H_ * 64) / 256, 256, 0, stream>>>(qkvf, Qb, Kb, Vt);

    flash_attn<<<dim3(S_ / 64, H_, B_), 128, 0, stream>>>(Qb, Kb, Vt, attnb);

    // out = attn @ w_o^T : M=4096, N=2048, K=2048 -> fp32 d_out
    gemm_bf16_nt<<<dim3(D_ / 128, (B_ * S_) / 128), 256, 0, stream>>>(
        attnb, wob, (float*)d_out, B_ * S_, D_, D_);
}
